// MambaMixer_75823352643749
// MI455X (gfx1250) — matrix operand order repaired and checked by reading, not compiled, not run on hardware
//
#include <hip/hip_runtime.h>
#include <math.h>

// ---------------- problem constants ----------------
#define HIDc   1024
#define INTERc 2048
#define STATEc 128
#define Hc     32
#define Pc     64
#define KCONV  4
#define Qc     256
#define CONVD  (INTERc + 2*STATEc)        // 2304
#define DINc   (2*INTERc + 2*STATEc + Hc) // 4384
#define Bc     2
#define Lc     2048
#define NCc    (Lc / Qc)                  // 8
#define BLc    (Bc * Lc)                  // 4096
#define NBC    (Bc * NCc)                 // 16

// ---------------- WMMA plumbing ----------------
typedef __attribute__((ext_vector_type(16))) __bf16 bf16x16;
typedef __attribute__((ext_vector_type(8)))  __bf16 bf16x8;
typedef __attribute__((ext_vector_type(2)))  __bf16 bf16x2;
typedef __attribute__((ext_vector_type(8)))  float  f32x8;
typedef __attribute__((ext_vector_type(2)))  float  f32x2;

// hardware float->bf16 conversion, packed pair (v_cvt_pk_bf16_f32 + ds_store_b32)
__device__ inline bf16x2 cvt2(float a, float b) {
    bf16x2 r;
    r[0] = (__bf16)a;
    r[1] = (__bf16)b;
    return r;
}

// A fragment (16x32 bf16, row-major LDS tile, 32-elem row stride):
// lane l: row = l%16, half = l/16; elems 0-7 -> K = 8*half+e, elems 8-15 -> K = 16+8*half+e
__device__ inline bf16x16 load_a16(const __bf16* row, int half) {
    bf16x8 lo = *(const bf16x8*)(row + 8 * half);
    bf16x8 hi = *(const bf16x8*)(row + 16 + 8 * half);
    return __builtin_shufflevector(lo, hi, 0,1,2,3,4,5,6,7,8,9,10,11,12,13,14,15);
}
// B fragment (32x16 bf16): staged n-major in LDS; lane l: col = l%16,
// elems 0-7 -> K = 8*half+e, elems 8-15 -> K = 16+8*half+e
__device__ inline bf16x16 load_b16(const __bf16* row, int half) {
    bf16x8 lo = *(const bf16x8*)(row + 8 * half);
    bf16x8 hi = *(const bf16x8*)(row + 16 + 8 * half);
    return __builtin_shufflevector(lo, hi, 0,1,2,3,4,5,6,7,8,9,10,11,12,13,14,15);
}
__device__ inline f32x8 wmma_bf16(bf16x16 a, bf16x16 b, f32x8 c) {
    return __builtin_amdgcn_wmma_f32_16x16x32_bf16(false, a, false, b, (short)0, c, false, false);
}

__device__ inline float sigmoidf_(float x) { return 1.f / (1.f + __expf(-x)); }

// ======================================================================
// Generic batched GEMM: C[m,n] = sum_k A[m,k] * B[n,k]   (A MxK, B NxK row-major)
// block tile: 128(m) x 64(n), 8 waves; wave = 16 rows x 64 cols = 4 WMMA accs.
// Software-pipelined: next K-step's tiles are register-prefetched behind WMMAs.
// grid = (M/128, ceil(N/64), nbatch). N guarded (in_proj N=4384 = 68*64+32).
// ======================================================================
__global__ void __launch_bounds__(256)
gemm_bf16_nt(const float* __restrict__ A, int lda, long long sA,
             const float* __restrict__ Bm, int ldb, long long sB,
             float* __restrict__ C, int ldc, long long sC, int Kdim, int N) {
    __shared__ alignas(32) __bf16 As[128][32];
    __shared__ alignas(32) __bf16 Bs[64][32];
    const int tid  = threadIdx.x;
    const int lane = tid & 31, wv = tid >> 5;
    const int r = lane & 15, half = lane >> 4;
    const int n0 = blockIdx.y * 64;
    const float* Ab = A  + (long long)blockIdx.z * sA + (long long)blockIdx.x * 128 * lda;
    const float* Bb = Bm + (long long)blockIdx.z * sB + (long long)n0 * ldb;
    float*       Cb = C  + (long long)blockIdx.z * sC + (long long)blockIdx.x * 128 * ldc + n0;
    f32x8 acc[4] = {};
    f32x2 ra[8], rb[4];
    auto load_tiles = [&](int k0) {
        #pragma unroll
        for (int i = 0; i < 8; ++i) {                  // 128x32 A tile, 2048 pairs
            int idx = tid + 256 * i;
            int rr = idx >> 4, c2 = (idx & 15) * 2;
            ra[i] = *(const f32x2*)&Ab[(long long)rr * lda + k0 + c2];
        }
        #pragma unroll
        for (int i = 0; i < 4; ++i) {                  // 64x32 B tile, 1024 pairs
            int idx = tid + 256 * i;
            int rr = idx >> 4, c2 = (idx & 15) * 2;
            f32x2 v = {0.f, 0.f};
            if (n0 + rr < N)
                v = *(const f32x2*)&Bb[(long long)rr * ldb + k0 + c2];
            rb[i] = v;
        }
    };
    load_tiles(0);
    for (int k0 = 0; k0 < Kdim; k0 += 32) {
        #pragma unroll
        for (int i = 0; i < 8; ++i) {
            int idx = tid + 256 * i;
            int rr = idx >> 4, c2 = (idx & 15) * 2;
            *(bf16x2*)&As[rr][c2] = cvt2(ra[i][0], ra[i][1]);
        }
        #pragma unroll
        for (int i = 0; i < 4; ++i) {
            int idx = tid + 256 * i;
            int rr = idx >> 4, c2 = (idx & 15) * 2;
            *(bf16x2*)&Bs[rr][c2] = cvt2(rb[i][0], rb[i][1]);
        }
        __syncthreads();
        if (k0 + 32 < Kdim) load_tiles(k0 + 32);       // prefetch behind WMMAs
        bf16x16 af = load_a16(&As[wv * 16 + r][0], half);
        #pragma unroll
        for (int f = 0; f < 4; ++f) {
            bf16x16 bfr = load_b16(&Bs[f * 16 + r][0], half);
            acc[f] = wmma_bf16(af, bfr, acc[f]);
        }
        __syncthreads();
    }
    #pragma unroll
    for (int f = 0; f < 4; ++f) {
        if (n0 + f * 16 + r < N) {
            #pragma unroll
            for (int v = 0; v < 8; ++v) {
                int m = wv * 16 + half * 8 + v;
                Cb[(long long)m * ldc + f * 16 + r] = acc[f][v];
            }
        }
    }
}

// ======================================================================
// Depthwise causal conv (K=4) + SiLU over zxbcdt[..., INTER:INTER+CONVD]
// ======================================================================
__global__ void conv_silu_kernel(const float* __restrict__ zxbcdt,
                                 const float* __restrict__ conv_w,
                                 const float* __restrict__ conv_b,
                                 float* __restrict__ xbc) {
    long long idx = (long long)blockIdx.x * 256 + threadIdx.x;
    if (idx >= (long long)BLc * CONVD) return;
    int ch = (int)(idx % CONVD);
    int bl = (int)(idx / CONVD);
    int b = bl / Lc, l = bl % Lc;
    float acc = conv_b[ch];
    #pragma unroll
    for (int t = 0; t < KCONV; ++t) {
        int lt = l - (KCONV - 1) + t;
        if (lt >= 0)
            acc += conv_w[ch * KCONV + t] *
                   zxbcdt[((long long)(b * Lc + lt)) * DINc + INTERc + ch];
    }
    xbc[idx] = acc * sigmoidf_(acc);
}

// dt = softplus(dt_raw + dt_bias), layout [bl][h]
__global__ void dt_kernel(const float* __restrict__ zxbcdt,
                          const float* __restrict__ dt_bias,
                          float* __restrict__ dtw) {
    int idx = blockIdx.x * 256 + threadIdx.x;
    if (idx >= BLc * Hc) return;
    int h = idx % Hc, bl = idx / Hc;
    float x = zxbcdt[(long long)bl * DINc + INTERc + CONVD + h] + dt_bias[h];
    dtw[idx] = (x > 20.f) ? x : log1pf(__expf(x));
}

// seg[bc][h][q] = cumsum_q(dt * A[h]) within each chunk
__global__ void seg_kernel(const float* __restrict__ dtw,
                           const float* __restrict__ A_log,
                           float* __restrict__ segw) {
    int idx = blockIdx.x * 256 + threadIdx.x;   // over NBC*H
    if (idx >= NBC * Hc) return;
    int h = idx % Hc, bc = idx / Hc;
    float Ah = -__expf(A_log[h]);
    float s = 0.f;
    for (int q = 0; q < Qc; ++q) {
        s += dtw[(bc * Qc + q) * Hc + h] * Ah;
        segw[((long long)bc * Hc + h) * Qc + q] = s;
    }
}

// ======================================================================
// states[bc,h,p,n] = sum_j x[j,h,p]*dt[j]*exp(seg_last - seg_j) * B[j,n]
// per-(bc,h) WMMA GEMM: M=P=64, N=STATE=128, K=Q=256. grid=(NBC, H)
// ======================================================================
__global__ void __launch_bounds__(256)
states_kernel(const float* __restrict__ xbc, const float* __restrict__ dtw,
              const float* __restrict__ segw, float* __restrict__ states) {
    const int bc = blockIdx.x, h = blockIdx.y;
    __shared__ alignas(32) __bf16 As[64][32];    // [p][j]
    __shared__ alignas(32) __bf16 Bs[128][32];   // [n][j]
    __shared__ float wj[32];
    const int tid = threadIdx.x, lane = tid & 31, wv = tid >> 5;
    const int r = lane & 15, half = lane >> 4;
    const int wm = wv & 3, wn = wv >> 2;
    const long long bcq = (long long)bc * Qc;
    const int hp = h * Pc;
    const float* seg = segw + ((long long)bc * Hc + h) * Qc;
    const float seglast = seg[Qc - 1];
    f32x8 acc[4] = {};
    float ax0[4], ax1[4], bx0[8], bx1[8];
    for (int j0 = 0; j0 < Qc; j0 += 32) {
        // issue all global loads first (one wait, many in flight)
        #pragma unroll
        for (int i = 0; i < 4; ++i) {            // A: 64x32, pairs along j
            int idx = tid + 256 * i;
            int p = idx >> 4, j2 = (idx & 15) * 2;
            ax0[i] = xbc[(bcq + j0 + j2)     * CONVD + hp + p];
            ax1[i] = xbc[(bcq + j0 + j2 + 1) * CONVD + hp + p];
        }
        #pragma unroll
        for (int i = 0; i < 8; ++i) {            // B: 128x32 (n-major)
            int idx = tid + 256 * i;
            int n = idx >> 4, j2 = (idx & 15) * 2;
            bx0[i] = xbc[(bcq + j0 + j2)     * CONVD + INTERc + n];
            bx1[i] = xbc[(bcq + j0 + j2 + 1) * CONVD + INTERc + n];
        }
        if (tid < 32) {
            int j = j0 + tid;
            wj[tid] = dtw[(bc * Qc + j) * Hc + h] * __expf(seglast - seg[j]);
        }
        __syncthreads();
        #pragma unroll
        for (int i = 0; i < 4; ++i) {
            int idx = tid + 256 * i;
            int p = idx >> 4, j2 = (idx & 15) * 2;
            *(bf16x2*)&As[p][j2] = cvt2(ax0[i] * wj[j2], ax1[i] * wj[j2 + 1]);
        }
        #pragma unroll
        for (int i = 0; i < 8; ++i) {
            int idx = tid + 256 * i;
            int n = idx >> 4, j2 = (idx & 15) * 2;
            *(bf16x2*)&Bs[n][j2] = cvt2(bx0[i], bx1[i]);
        }
        __syncthreads();
        bf16x16 af = load_a16(&As[wm * 16 + r][0], half);
        #pragma unroll
        for (int f = 0; f < 4; ++f) {
            bf16x16 bfr = load_b16(&Bs[wn * 64 + f * 16 + r][0], half);
            acc[f] = wmma_bf16(af, bfr, acc[f]);
        }
        __syncthreads();
    }
    float* out = states + ((long long)(bc * Hc + h)) * Pc * STATEc;
    #pragma unroll
    for (int f = 0; f < 4; ++f)
        #pragma unroll
        for (int v = 0; v < 8; ++v) {
            int p = wm * 16 + half * 8 + v;
            int n = wn * 64 + f * 16 + r;
            out[(long long)p * STATEc + n] = acc[f][v];
        }
}

// inter-chunk serial scan: prev[bc] = carry before chunk bc
__global__ void scan_kernel(const float* __restrict__ states,
                            const float* __restrict__ segw,
                            float* __restrict__ prev) {
    const int HPS = Hc * Pc * STATEc;
    int idx = blockIdx.x * 256 + threadIdx.x;    // over B*HPS
    if (idx >= Bc * HPS) return;
    int b = idx / HPS, rr = idx % HPS;
    int h = rr / (Pc * STATEc);
    float carry = 0.f;
    for (int c = 0; c < NCc; ++c) {
        int bc = b * NCc + c;
        prev[(long long)bc * HPS + rr] = carry;
        float dec = __expf(segw[((long long)bc * Hc + h) * Qc + (Qc - 1)]);
        carry = carry * dec + states[(long long)bc * HPS + rr];
    }
}

// ======================================================================
// Fused Yd + Yo + D*x:
//   Yd[i,p] = sum_j (CB[i,j]*exp(seg_i-seg_j)[i>=j]) * x[j,p]*dt[j]
//   Yo[i,p] = sum_n (C[i,n]*exp(seg_i)) * prev[p,n]
// per-(bc,h) tile of 128 rows; grid=(Q/128, H, NBC)
// ======================================================================
__global__ void __launch_bounds__(256)
ydyo_kernel(const float* __restrict__ xbc, const float* __restrict__ dtw,
            const float* __restrict__ segw, const float* __restrict__ CBw,
            const float* __restrict__ prev, const float* __restrict__ Dvec,
            float* __restrict__ yw) {
    const int mt = blockIdx.x, h = blockIdx.y, bc = blockIdx.z;
    __shared__ alignas(32) __bf16 As[128][32];
    __shared__ alignas(32) __bf16 Bs[64][32];
    __shared__ float segi[128];
    __shared__ float segj[32];
    __shared__ float dtj[32];
    const int tid = threadIdx.x, lane = tid & 31, wv = tid >> 5;
    const int r = lane & 15, half = lane >> 4;
    const int i0 = mt * 128;
    const float* seg = segw + ((long long)bc * Hc + h) * Qc;
    if (tid < 128) segi[tid] = seg[i0 + tid];
    const long long rowbase = (long long)bc * Qc;   // global token row
    const int hp = h * Pc;
    f32x8 acc[4] = {};
    // ---- phase 1: Yd, K over j (Q) ----
    for (int j0 = 0; j0 < Qc; j0 += 32) {
        f32x2 cb[8];
        float bx0[4], bx1[4];
        #pragma unroll
        for (int i = 0; i < 8; ++i) {            // CB loads (lower-tri only)
            int idx = tid + 256 * i;
            int ii = idx >> 4, j2 = (idx & 15) * 2;
            int gi = i0 + ii, gj = j0 + j2;
            f32x2 v = {0.f, 0.f};
            if (gi >= gj)
                v = *(const f32x2*)&CBw[(rowbase + gi) * Qc + gj];
            cb[i] = v;
        }
        #pragma unroll
        for (int i = 0; i < 4; ++i) {            // x loads for xdt tile
            int idx = tid + 256 * i;
            int p = idx >> 4, j2 = (idx & 15) * 2;
            bx0[i] = xbc[(rowbase + j0 + j2)     * CONVD + hp + p];
            bx1[i] = xbc[(rowbase + j0 + j2 + 1) * CONVD + hp + p];
        }
        if (tid < 32) {
            segj[tid] = seg[j0 + tid];
            dtj[tid]  = dtw[(rowbase + j0 + tid) * Hc + h];
        }
        __syncthreads();
        #pragma unroll
        for (int i = 0; i < 8; ++i) {            // masked CB*L tile 128x32
            int idx = tid + 256 * i;
            int ii = idx >> 4, j2 = (idx & 15) * 2;
            int gi = i0 + ii, gj = j0 + j2;
            float a0 = 0.f, a1 = 0.f;
            if (gi >= gj)     a0 = cb[i][0] * __expf(segi[ii] - segj[j2]);
            if (gi >= gj + 1) a1 = cb[i][1] * __expf(segi[ii] - segj[j2 + 1]);
            *(bf16x2*)&As[ii][j2] = cvt2(a0, a1);
        }
        #pragma unroll
        for (int i = 0; i < 4; ++i) {            // xdt tile 64(p)x32(j)
            int idx = tid + 256 * i;
            int p = idx >> 4, j2 = (idx & 15) * 2;
            *(bf16x2*)&Bs[p][j2] = cvt2(bx0[i] * dtj[j2], bx1[i] * dtj[j2 + 1]);
        }
        __syncthreads();
        bf16x16 af = load_a16(&As[wv * 16 + r][0], half);
        #pragma unroll
        for (int f = 0; f < 4; ++f) {
            bf16x16 bfr = load_b16(&Bs[f * 16 + r][0], half);
            acc[f] = wmma_bf16(af, bfr, acc[f]);
        }
        __syncthreads();
    }
    // ---- phase 2: Yo, K over n (STATE) ----
    const float* prevb = prev + ((long long)(bc * Hc + h)) * Pc * STATEc;
    for (int n0 = 0; n0 < STATEc; n0 += 32) {
        f32x2 cv[8], pv[4];
        #pragma unroll
        for (int i = 0; i < 8; ++i) {            // C loads
            int idx = tid + 256 * i;
            int ii = idx >> 4, n2 = (idx & 15) * 2;
            cv[i] = *(const f32x2*)&xbc[(rowbase + i0 + ii) * CONVD
                                        + INTERc + STATEc + n0 + n2];
        }
        #pragma unroll
        for (int i = 0; i < 4; ++i) {            // prev loads
            int idx = tid + 256 * i;
            int p = idx >> 4, n2 = (idx & 15) * 2;
            pv[i] = *(const f32x2*)&prevb[(long long)p * STATEc + n0 + n2];
        }
        #pragma unroll
        for (int i = 0; i < 8; ++i) {            // C * exp(seg_i) tile
            int idx = tid + 256 * i;
            int ii = idx >> 4, n2 = (idx & 15) * 2;
            float e = __expf(segi[ii]);
            *(bf16x2*)&As[ii][n2] = cvt2(cv[i][0] * e, cv[i][1] * e);
        }
        #pragma unroll
        for (int i = 0; i < 4; ++i) {            // prev tile 64(p)x32(n)
            int idx = tid + 256 * i;
            int p = idx >> 4, n2 = (idx & 15) * 2;
            *(bf16x2*)&Bs[p][n2] = cvt2(pv[i][0], pv[i][1]);
        }
        __syncthreads();
        bf16x16 af = load_a16(&As[wv * 16 + r][0], half);
        #pragma unroll
        for (int f = 0; f < 4; ++f) {
            bf16x16 bfr = load_b16(&Bs[f * 16 + r][0], half);
            acc[f] = wmma_bf16(af, bfr, acc[f]);
        }
        __syncthreads();
    }
    // ---- epilogue: + D[h]*x, write y ----
    const float Dh = Dvec[h];
    #pragma unroll
    for (int f = 0; f < 4; ++f)
        #pragma unroll
        for (int v = 0; v < 8; ++v) {
            int ii = wv * 16 + half * 8 + v;
            int p  = f * 16 + r;
            long long row = rowbase + i0 + ii;
            float xv = xbc[row * CONVD + hp + p];
            yw[row * INTERc + hp + p] = acc[f][v] + Dh * xv;
        }
}

// gated RMSNorm per token: y = (y*silu(z)) * rsqrt(mean(sq)+eps) * w
__global__ void __launch_bounds__(256)
gatenorm_kernel(const float* __restrict__ zxbcdt, const float* __restrict__ nw,
                float* __restrict__ yw) {
    const int bl = blockIdx.x;
    const float* z = zxbcdt + (long long)bl * DINc;
    float* y = yw + (long long)bl * INTERc;
    float ss = 0.f;
    for (int i = threadIdx.x; i < INTERc; i += 256) {
        float zv = z[i];
        float v = y[i] * (zv * sigmoidf_(zv));
        y[i] = v;
        ss += v * v;
    }
    __shared__ float red[256];
    red[threadIdx.x] = ss;
    __syncthreads();
    for (int s = 128; s > 0; s >>= 1) {
        if (threadIdx.x < s) red[threadIdx.x] += red[threadIdx.x + s];
        __syncthreads();
    }
    float rs = rsqrtf(red[0] / (float)INTERc + 1e-6f);
    for (int i = threadIdx.x; i < INTERc; i += 256)
        y[i] = y[i] * rs * nw[i];
}

// ======================================================================
extern "C" void kernel_launch(void* const* d_in, const int* in_sizes, int n_in,
                              void* d_out, int out_size, void* d_ws, size_t ws_size,
                              hipStream_t stream) {
    const float* hidden     = (const float*)d_in[0];
    const float* in_proj_w  = (const float*)d_in[1];
    const float* conv_w     = (const float*)d_in[2];
    const float* conv_b     = (const float*)d_in[3];
    const float* dt_bias    = (const float*)d_in[4];
    const float* A_log      = (const float*)d_in[5];
    const float* Dvec       = (const float*)d_in[6];
    const float* norm_w     = (const float*)d_in[7];
    const float* out_proj_w = (const float*)d_in[8];
    float* out = (float*)d_out;

    float* ws = (float*)d_ws;                         // ~174 MB total
    float* zxbcdt = ws;  ws += (size_t)BLc * DINc;          // 17.96M f
    float* xbc    = ws;  ws += (size_t)BLc * CONVD;         //  9.44M f
    float* dtw    = ws;  ws += (size_t)BLc * Hc;            //  0.13M f
    float* segw   = ws;  ws += (size_t)BLc * Hc;            //  0.13M f
    float* CBw    = ws;  ws += (size_t)NBC * Qc * Qc;       //  1.05M f
    float* stw    = ws;  ws += (size_t)NBC * Hc * Pc * STATEc; // 4.19M f
    float* prevw  = ws;  ws += (size_t)NBC * Hc * Pc * STATEc; // 4.19M f
    float* yw     = ws;  ws += (size_t)BLc * INTERc;        //  8.39M f
    (void)ws; (void)ws_size; (void)in_sizes; (void)n_in; (void)out_size;

    // 1) in_proj: zxbcdt[4096,4384] = hidden[4096,1024] @ W[4384,1024]^T
    gemm_bf16_nt<<<dim3(BLc / 128, (DINc + 63) / 64, 1), 256, 0, stream>>>(
        hidden, HIDc, 0LL, in_proj_w, HIDc, 0LL, zxbcdt, DINc, 0LL, HIDc, DINc);

    // 2) depthwise conv + SiLU
    conv_silu_kernel<<<(BLc * CONVD) / 256, 256, 0, stream>>>(zxbcdt, conv_w, conv_b, xbc);

    // 3) dt = softplus, seg = cumsum(dt*A)
    dt_kernel<<<(BLc * Hc) / 256, 256, 0, stream>>>(zxbcdt, dt_bias, dtw);
    seg_kernel<<<(NBC * Hc + 255) / 256, 256, 0, stream>>>(dtw, A_log, segw);

    // 4) CB[bc,256,256] = C[256,128] @ B[256,128]^T  (batched x16)
    gemm_bf16_nt<<<dim3(Qc / 128, Qc / 64, NBC), 256, 0, stream>>>(
        xbc + INTERc + STATEc, CONVD, (long long)Qc * CONVD,
        xbc + INTERc,          CONVD, (long long)Qc * CONVD,
        CBw, Qc, (long long)Qc * Qc, STATEc, Qc);

    // 5) per-chunk states, then serial inter-chunk scan
    states_kernel<<<dim3(NBC, Hc), 256, 0, stream>>>(xbc, dtw, segw, stw);
    scan_kernel<<<(Bc * Hc * Pc * STATEc) / 256, 256, 0, stream>>>(stw, segw, prevw);

    // 6) fused Yd + Yo + D*x
    ydyo_kernel<<<dim3(Qc / 128, Hc, NBC), 256, 0, stream>>>(
        xbc, dtw, segw, CBw, prevw, Dvec, yw);

    // 7) gate + RMSNorm
    gatenorm_kernel<<<BLc, 256, 0, stream>>>(zxbcdt, norm_w, yw);

    // 8) out_proj: out[4096,1024] = y[4096,2048] @ Wout[1024,2048]^T
    gemm_bf16_nt<<<dim3(BLc / 128, HIDc / 64, 1), 256, 0, stream>>>(
        yw, INTERc, 0LL, out_proj_w, INTERc, 0LL, out, HIDc, 0LL, INTERc, HIDc);
}
